// MoE_91096256348853
// MI455X (gfx1250) — matrix operand order repaired and checked by reading, not compiled, not run on hardware
//
#include <hip/hip_runtime.h>
#include <math.h>

#define BB 2048
#define DD 1024
#define HH 2048
#define OO 512
#define EE 8
#define SS (BB * 2)   // total dispatch slots = B * TOP_K = 4096 (exact)
#define LDA 40        // padded LDS row stride (bf16 elems) for K=32 tiles -> conflict-free b128

#define TILE_ELEMS (128 * LDA)          // elems per tile buffer (fc1)
#define TILE_BYTES (TILE_ELEMS * 2)     // 10240 B

typedef __attribute__((ext_vector_type(16))) __bf16          bf16x16;
typedef __attribute__((ext_vector_type(8)))  unsigned short  u16x8;
typedef __attribute__((ext_vector_type(8)))  float           f32x8;

union Frag { bf16x16 v; u16x8 h[2]; };

// wave32 xor-shuffle via ds_swizzle (group-of-32: and=31, or=0, xor=M)
#define SWZ(v, M) __int_as_float(__builtin_amdgcn_ds_swizzle(__float_as_int(v), (31 | ((M) << 10))))

__device__ __forceinline__ unsigned short f2bf(float f) {
  __bf16 b = (__bf16)f;
  return __builtin_bit_cast(unsigned short, b);
}

// CDNA5 async copy: global -> LDS, 16B per lane, tracked by ASYNCcnt
__device__ __forceinline__ void async_ld16(unsigned lds_byte_off, const unsigned short* g) {
  asm volatile("global_load_async_to_lds_b128 %0, %1, off"
               :: "v"(lds_byte_off), "v"((unsigned long long)(uintptr_t)g)
               : "memory");
}

// ---------------- gating ----------------
__global__ __launch_bounds__(256) void zero_k(int* cnt) {
  if (threadIdx.x < EE) cnt[threadIdx.x] = 0;
}

__global__ __launch_bounds__(256) void gate_k(const float* __restrict__ x,
                                              const float* __restrict__ wg,
                                              int* __restrict__ top_idx,
                                              float* __restrict__ top_gate,
                                              int* __restrict__ cnt) {
  int wv = threadIdx.x >> 5, lane = threadIdx.x & 31;
  int b = blockIdx.x * 8 + wv;
  float acc[EE];
#pragma unroll
  for (int e = 0; e < EE; ++e) acc[e] = 0.f;
  for (int d = lane; d < DD; d += 32) {
    float xv = x[(size_t)b * DD + d];
    const float* wr = wg + (size_t)d * EE;
#pragma unroll
    for (int e = 0; e < EE; ++e) acc[e] = fmaf(xv, wr[e], acc[e]);
  }
#pragma unroll
  for (int e = 0; e < EE; ++e) {
    float v = acc[e];
    v += SWZ(v, 16); v += SWZ(v, 8); v += SWZ(v, 4); v += SWZ(v, 2); v += SWZ(v, 1);
    acc[e] = v;
  }
  if (lane == 0) {
    int i0 = 0; float v0 = acc[0];
#pragma unroll
    for (int e = 1; e < EE; ++e) if (acc[e] > v0) { v0 = acc[e]; i0 = e; }
    int i1 = (i0 == 0) ? 1 : 0; float v1 = acc[i1];
#pragma unroll
    for (int e = 0; e < EE; ++e) if (e != i0 && acc[e] > v1) { v1 = acc[e]; i1 = e; }
    float e1 = expf(v1 - v0);
    float inv = 1.f / (1.f + e1);
    top_idx[2 * b]     = i0;  top_idx[2 * b + 1]  = i1;
    top_gate[2 * b]    = inv; top_gate[2 * b + 1] = e1 * inv;
    atomicAdd(&cnt[i0], 1);   atomicAdd(&cnt[i1], 1);
  }
}

__global__ void scan_k(const int* __restrict__ cnt, int* __restrict__ offs, int* __restrict__ cursor) {
  if (threadIdx.x == 0) {
    int s = 0;
    for (int e = 0; e < EE; ++e) { offs[e] = s; cursor[e] = s; s += cnt[e]; }
    offs[EE] = s;
  }
}

__global__ __launch_bounds__(256) void assign_k(const int* __restrict__ top_idx,
                                                const float* __restrict__ top_gate,
                                                int* __restrict__ cursor,
                                                int* __restrict__ slot_of,
                                                int* __restrict__ row_token,
                                                float* __restrict__ slot_gate) {
  int b = blockIdx.x * 256 + threadIdx.x;
  if (b >= BB) return;
#pragma unroll
  for (int k = 0; k < 2; ++k) {
    int e = top_idx[2 * b + k];
    int slot = atomicAdd(&cursor[e], 1);
    slot_of[2 * b + k] = slot;
    row_token[slot] = b;
    slot_gate[slot] = top_gate[2 * b + k];
  }
}

// ---------------- precision prep ----------------
__global__ __launch_bounds__(256) void cvt_k(const float* __restrict__ in,
                                             unsigned short* __restrict__ out, int n) {
  int i = blockIdx.x * 256 + threadIdx.x;
  if (i < n) out[i] = f2bf(in[i]);
}

// per-expert [R][C] f32 -> [C][R] bf16 (N-major weights so B-fragments load contiguous K)
__global__ __launch_bounds__(256) void trcvt_k(const float* __restrict__ in,
                                               unsigned short* __restrict__ out,
                                               int R, int C) {
  __shared__ float t[32][33];
  int e = blockIdx.z;
  const float* src = in + (size_t)e * R * C;
  unsigned short* dst = out + (size_t)e * R * C;
  int c0 = blockIdx.x * 32, r0 = blockIdx.y * 32;
  int tx = threadIdx.x, ty = threadIdx.y;
#pragma unroll
  for (int i = ty; i < 32; i += 8) t[i][tx] = src[(size_t)(r0 + i) * C + c0 + tx];
  __syncthreads();
#pragma unroll
  for (int i = ty; i < 32; i += 8) dst[(size_t)(c0 + i) * R + r0 + tx] = f2bf(t[tx][i]);
}

// ---------------- GEMM1: h = gelu(x @ fc1_w + b1), gathered per expert ----------------
// Async path: GLOBAL_LOAD_ASYNC_TO_LDS_B128 + s_wait_asynccnt, quad-buffered LDS
// (>=3 needed for the issue-ahead window; 4 divides the trip count so unroll-4
// folds buffer offsets to compile-time immediates), one barrier per K-step.
__global__ __launch_bounds__(256) void fc1_k(const unsigned short* __restrict__ xb,
                                             const unsigned short* __restrict__ w1t, // [E][H][D]
                                             const float* __restrict__ b1,
                                             const int* __restrict__ offs,
                                             const int* __restrict__ row_token,
                                             unsigned short* __restrict__ hbuf) {    // [S][H]
  int e = blockIdx.z;
  int base = offs[e], cntE = offs[e + 1] - base;
  int mBase = blockIdx.x * 128;
  if (mBase >= cntE) return;
  int nBase = blockIdx.y * 128;

  __shared__ __align__(16) unsigned short As[4][TILE_ELEMS];
  __shared__ __align__(16) unsigned short Bs[4][TILE_ELEMS];
  __shared__ int tok[128];

  int tid = threadIdx.x;
  if (tid < 128) {
    int r = mBase + tid;
    tok[tid] = (r < cntE) ? row_token[base + r] : 0;
  }
  __syncthreads();

  int lane = tid & 31, wv = tid >> 5;
  int wm = (wv >> 2) * 64, wn = (wv & 3) * 32;
  int l15 = lane & 15, hi = lane >> 4;
  int lrow = tid >> 1, lhalf = tid & 1;

  // loop-invariant per-thread source pointers and LDS byte offsets
  const unsigned short* aSrc = xb + (size_t)tok[lrow] * DD + lhalf * 16;
  const unsigned short* bSrc = w1t + ((size_t)e * HH + nBase + lrow) * DD + lhalf * 16;
  unsigned myOff = (unsigned)(lrow * (LDA * 2) + lhalf * 32);
  unsigned asB = (unsigned)(uintptr_t)(&As[0][0]) + myOff;   // LDS aperture: low 32 bits = LDS offset
  unsigned bsB = (unsigned)(uintptr_t)(&Bs[0][0]) + myOff;

  f32x8 acc[4][2];
#pragma unroll
  for (int mi = 0; mi < 4; ++mi)
#pragma unroll
    for (int ni = 0; ni < 2; ++ni)
#pragma unroll
      for (int j = 0; j < 8; ++j) acc[mi][ni][j] = 0.f;

  // prologue: tile 0 -> buffer 0 (4 async b128 per thread per tile)
  async_ld16(asB,      aSrc);
  async_ld16(asB + 16, aSrc + 8);
  async_ld16(bsB,      bSrc);
  async_ld16(bsB + 16, bSrc + 8);

#pragma unroll 4
  for (int kk = 0; kk < DD; kk += 32) {
    int p  = (kk >> 5) & 3;        // compile-time per unrolled body
    int pn = (p + 1) & 3;
    if (kk + 32 < DD) {
      unsigned ao = asB + (unsigned)pn * TILE_BYTES;
      unsigned bo = bsB + (unsigned)pn * TILE_BYTES;
      async_ld16(ao,      aSrc + kk + 32);
      async_ld16(ao + 16, aSrc + kk + 40);
      async_ld16(bo,      bSrc + kk + 32);
      async_ld16(bo + 16, bSrc + kk + 40);
      // in-order completion: <=4 outstanding means the current tile has landed
      asm volatile("s_wait_asynccnt 4" ::: "memory");
    } else {
      asm volatile("s_wait_asynccnt 0" ::: "memory");
    }
    __syncthreads();

    const unsigned short* Ac = &As[p][0];
    const unsigned short* Bc = &Bs[p][0];
    Frag af[4], bf[2];
#pragma unroll
    for (int mi = 0; mi < 4; ++mi) {           // A frag: lanes<16 K0..7,K16..23 ; lanes>=16 +8
      int r = wm + mi * 16 + l15;
      af[mi].h[0] = *(const u16x8*)(Ac + r * LDA + hi * 8);
      af[mi].h[1] = *(const u16x8*)(Ac + r * LDA + hi * 8 + 16);
    }
#pragma unroll
    for (int ni = 0; ni < 2; ++ni) {           // B frag: lanes<16 K0..7,K16..23 ; lanes>=16 +8
      int c = wn + ni * 16 + l15;
      bf[ni].h[0] = *(const u16x8*)(Bc + c * LDA + hi * 8);
      bf[ni].h[1] = *(const u16x8*)(Bc + c * LDA + hi * 8 + 16);
    }
#pragma unroll
    for (int mi = 0; mi < 4; ++mi)
#pragma unroll
      for (int ni = 0; ni < 2; ++ni)
        acc[mi][ni] = __builtin_amdgcn_wmma_f32_16x16x32_bf16(
            false, af[mi].v, false, bf[ni].v, (short)0, acc[mi][ni], false, false);
  }

#pragma unroll
  for (int mi = 0; mi < 4; ++mi)
#pragma unroll
    for (int ni = 0; ni < 2; ++ni) {
      int c = nBase + wn + ni * 16 + l15;
      float bias = b1[(size_t)e * HH + c];
#pragma unroll
      for (int j = 0; j < 8; ++j) {
        int r = mBase + wm + mi * 16 + hi * 8 + j;
        if (r < cntE) {
          float v = acc[mi][ni][j] + bias;
          float g = 0.5f * v * (1.0f + erff(v * 0.70710678118654752f));  // exact gelu
          hbuf[(size_t)(base + r) * HH + c] = f2bf(g);
        }
      }
    }
}

// ---------------- GEMM2: logits2 = h @ fc2_w + b2 ; probs = gate * softmax(row) ----------------
__global__ __launch_bounds__(256) void fc2_k(const unsigned short* __restrict__ hbuf, // [S][H]
                                             const unsigned short* __restrict__ w2t,  // [E][O][H]
                                             const float* __restrict__ b2,
                                             const int* __restrict__ offs,
                                             const float* __restrict__ slot_gate,
                                             float* __restrict__ pbuf) {              // [S][O]
  int e = blockIdx.y;
  int base = offs[e], cntE = offs[e + 1] - base;
  int mBase = blockIdx.x * 32;
  if (mBase >= cntE) return;

  __shared__ __align__(16) unsigned short As[32 * LDA];
  __shared__ __align__(16) unsigned short Bs[OO * LDA];
  __shared__ float red[32][8];
  __shared__ float rowz[32];
  __shared__ float gar[32];

  int tid = threadIdx.x;
  if (tid < 32) {
    int s = base + mBase + tid;
    gar[tid] = slot_gate[s < SS ? s : (SS - 1)];
  }
  int lane = tid & 31, wv = tid >> 5;
  int l15 = lane & 15, hi = lane >> 4;
  int cBase = wv * 64;  // 8 waves cover all O=512

  f32x8 acc[2][4];
#pragma unroll
  for (int mi = 0; mi < 2; ++mi)
#pragma unroll
    for (int ni = 0; ni < 4; ++ni)
#pragma unroll
      for (int j = 0; j < 8; ++j) acc[mi][ni][j] = 0.f;

  const unsigned short* wbase = w2t + (size_t)e * OO * HH;
  int arow = tid >> 1, ahalf = tid & 1;
  int aslot = base + mBase + arow;
  if (aslot >= SS) aslot = SS - 1;
  const unsigned short* asrc = hbuf + (size_t)aslot * HH + ahalf * 16;

  for (int kk = 0; kk < HH; kk += 32) {
    u16x8 a0, a1;
#pragma unroll
    for (int q = 0; q < 8; ++q) { a0[q] = 0; a1[q] = 0; }
    if (tid < 64) {
      const u16x8* pa = (const u16x8*)(asrc + kk);
      a0 = pa[0]; a1 = pa[1];
    }
    u16x8 br[4][2];
#pragma unroll
    for (int i = 0; i < 4; ++i) {
      int idx = tid + i * 256;
      int n = idx >> 1, h = idx & 1;
      const u16x8* pb = (const u16x8*)(wbase + (size_t)n * HH + kk + h * 16);
      br[i][0] = pb[0]; br[i][1] = pb[1];
    }
    if (kk + 32 < HH)
      __builtin_prefetch(wbase + (size_t)(tid >> 1) * HH + kk + 32, 0, 3);

    __syncthreads();
    if (tid < 64) {
      *(u16x8*)(As + arow * LDA + ahalf * 16)     = a0;
      *(u16x8*)(As + arow * LDA + ahalf * 16 + 8) = a1;
    }
#pragma unroll
    for (int i = 0; i < 4; ++i) {
      int idx = tid + i * 256;
      int n = idx >> 1, h = idx & 1;
      *(u16x8*)(Bs + n * LDA + h * 16)     = br[i][0];
      *(u16x8*)(Bs + n * LDA + h * 16 + 8) = br[i][1];
    }
    __syncthreads();

    Frag af[2], bf[4];
#pragma unroll
    for (int mi = 0; mi < 2; ++mi) {
      int r = mi * 16 + l15;
      af[mi].h[0] = *(const u16x8*)(As + r * LDA + hi * 8);
      af[mi].h[1] = *(const u16x8*)(As + r * LDA + hi * 8 + 16);
    }
#pragma unroll
    for (int ni = 0; ni < 4; ++ni) {
      int c = cBase + ni * 16 + l15;
      bf[ni].h[0] = *(const u16x8*)(Bs + c * LDA + hi * 8);
      bf[ni].h[1] = *(const u16x8*)(Bs + c * LDA + hi * 8 + 16);
    }
#pragma unroll
    for (int mi = 0; mi < 2; ++mi)
#pragma unroll
      for (int ni = 0; ni < 4; ++ni)
        acc[mi][ni] = __builtin_amdgcn_wmma_f32_16x16x32_bf16(
            false, af[mi].v, false, bf[ni].v, (short)0, acc[mi][ni], false, false);
  }

  // bias
#pragma unroll
  for (int ni = 0; ni < 4; ++ni) {
    float bias = b2[(size_t)e * OO + cBase + ni * 16 + l15];
#pragma unroll
    for (int mi = 0; mi < 2; ++mi)
#pragma unroll
      for (int j = 0; j < 8; ++j) acc[mi][ni][j] += bias;
  }
  // row max: in-wave (64 cols) via xor-shuffle inside 16-lane halves, then cross-wave via LDS
  float lm[2][8];
#pragma unroll
  for (int mi = 0; mi < 2; ++mi)
#pragma unroll
    for (int j = 0; j < 8; ++j) {
      float m = fmaxf(fmaxf(acc[mi][0][j], acc[mi][1][j]), fmaxf(acc[mi][2][j], acc[mi][3][j]));
      m = fmaxf(m, SWZ(m, 1)); m = fmaxf(m, SWZ(m, 2)); m = fmaxf(m, SWZ(m, 4)); m = fmaxf(m, SWZ(m, 8));
      lm[mi][j] = m;
    }
  if (l15 == 0)
#pragma unroll
    for (int mi = 0; mi < 2; ++mi)
#pragma unroll
      for (int j = 0; j < 8; ++j) red[mi * 16 + hi * 8 + j][wv] = lm[mi][j];
  __syncthreads();
  if (tid < 32) {
    float m = red[tid][0];
#pragma unroll
    for (int w = 1; w < 8; ++w) m = fmaxf(m, red[tid][w]);
    rowz[tid] = m;
  }
  __syncthreads();
  // row sumexp
#pragma unroll
  for (int mi = 0; mi < 2; ++mi)
#pragma unroll
    for (int j = 0; j < 8; ++j) {
      float rm = rowz[mi * 16 + hi * 8 + j];
      float s = expf(acc[mi][0][j] - rm) + expf(acc[mi][1][j] - rm) +
                expf(acc[mi][2][j] - rm) + expf(acc[mi][3][j] - rm);
      s += SWZ(s, 1); s += SWZ(s, 2); s += SWZ(s, 4); s += SWZ(s, 8);
      lm[mi][j] = s;
    }
  if (l15 == 0)
#pragma unroll
    for (int mi = 0; mi < 2; ++mi)
#pragma unroll
      for (int j = 0; j < 8; ++j) red[mi * 16 + hi * 8 + j][wv] = lm[mi][j];
  __syncthreads();
  if (tid < 32) {
    float s = 0.f;
#pragma unroll
    for (int w = 0; w < 8; ++w) s += red[tid][w];
    rowz[tid] = rowz[tid] + logf(s);  // logZ
  }
  __syncthreads();
  // probs = gate * exp(v - logZ)
#pragma unroll
  for (int mi = 0; mi < 2; ++mi)
#pragma unroll
    for (int j = 0; j < 8; ++j) {
      int rl = mi * 16 + hi * 8 + j;
      int r = mBase + rl;
      if (r < cntE) {
        float lz = rowz[rl];
        float g = gar[rl];
#pragma unroll
        for (int ni = 0; ni < 4; ++ni) {
          int c = cBase + ni * 16 + l15;
          pbuf[(size_t)(base + r) * OO + c] = expf(acc[mi][ni][j] - lz) * g;
        }
      }
    }
}

// ---------------- combine: out = log(p0 + p1), clamp 0 -> eps ----------------
__global__ __launch_bounds__(256) void combine_k(const float* __restrict__ pbuf,
                                                 const int* __restrict__ slot_of,
                                                 float* __restrict__ out) {
  int idx = blockIdx.x * 256 + threadIdx.x;
  int b = idx >> 9, o = idx & (OO - 1);
  float c = pbuf[(size_t)slot_of[2 * b] * OO + o] + pbuf[(size_t)slot_of[2 * b + 1] * OO + o];
  if (c == 0.0f) c = 2.2204460492503131e-16f;  // np.finfo(float64).eps
  out[idx] = logf(c);
}

extern "C" void kernel_launch(void* const* d_in, const int* in_sizes, int n_in,
                              void* d_out, int out_size, void* d_ws, size_t ws_size,
                              hipStream_t stream) {
  (void)in_sizes; (void)n_in; (void)out_size; (void)ws_size;
  const float* x    = (const float*)d_in[0];
  const float* wg   = (const float*)d_in[1];
  const float* fc1w = (const float*)d_in[2];
  const float* fc1b = (const float*)d_in[3];
  const float* fc2w = (const float*)d_in[4];
  const float* fc2b = (const float*)d_in[5];
  float* out = (float*)d_out;

  char* w = (char*)d_ws;
  size_t off = 0;
  auto alloc = [&](size_t bytes) {
    void* p = w + off;
    off = (off + bytes + 255) & ~(size_t)255;
    return p;
  };
  int*   cnt      = (int*)alloc(EE * 4);
  int*   offs     = (int*)alloc((EE + 1) * 4);
  int*   cursor   = (int*)alloc(EE * 4);
  int*   top_idx  = (int*)alloc((size_t)BB * 2 * 4);
  float* top_gate = (float*)alloc((size_t)BB * 2 * 4);
  int*   slot_of  = (int*)alloc((size_t)BB * 2 * 4);
  int*   row_tok  = (int*)alloc((size_t)SS * 4);
  float* sgate    = (float*)alloc((size_t)SS * 4);
  unsigned short* xb  = (unsigned short*)alloc((size_t)BB * DD * 2);
  unsigned short* w1t = (unsigned short*)alloc((size_t)EE * HH * DD * 2);
  unsigned short* w2t = (unsigned short*)alloc((size_t)EE * OO * HH * 2);
  unsigned short* hb  = (unsigned short*)alloc((size_t)SS * HH * 2);
  float* pbuf = (float*)alloc((size_t)SS * OO * 4);

  zero_k<<<1, 32, 0, stream>>>(cnt);
  cvt_k<<<(BB * DD) / 256, 256, 0, stream>>>(x, xb, BB * DD);
  trcvt_k<<<dim3(HH / 32, DD / 32, EE), dim3(32, 8), 0, stream>>>(fc1w, w1t, DD, HH);
  trcvt_k<<<dim3(OO / 32, HH / 32, EE), dim3(32, 8), 0, stream>>>(fc2w, w2t, HH, OO);
  gate_k<<<BB / 8, 256, 0, stream>>>(x, wg, top_idx, top_gate, cnt);
  scan_k<<<1, 32, 0, stream>>>(cnt, offs, cursor);
  assign_k<<<BB / 256, 256, 0, stream>>>(top_idx, top_gate, cursor, slot_of, row_tok, sgate);
  fc1_k<<<dim3(BB / 128, HH / 128, EE), 256, 0, stream>>>(xb, w1t, fc1b, offs, row_tok, hb);
  fc2_k<<<dim3(BB / 32, EE), 256, 0, stream>>>(hb, w2t, fc2b, offs, sgate, pbuf);
  combine_k<<<(BB * OO) / 256, 256, 0, stream>>>(pbuf, slot_of, out);
}
